// EncoderBlock_58531814310147
// MI455X (gfx1250) — compile-verified, matrix operand order checked by reading, not run on hardware
//
#include <hip/hip_runtime.h>

typedef __attribute__((ext_vector_type(16))) __bf16 v16bf;
typedef __attribute__((ext_vector_type(8)))  __bf16 bf16x8;
typedef __attribute__((ext_vector_type(8)))  float  v8f;

// ---------------- problem sizes ----------------
constexpr int cB  = 8;
constexpr int cT  = 8;
constexpr int cN  = 196;
constexpr int cD  = 768;
constexpr int cH  = 12;
constexpr int cDH = 64;
constexpr int cS  = cT * cN + 1;        // 1569
constexpr int D3  = 3 * cD;             // 2304
constexpr int LS  = cN + 1;             // 197 (spatial seq len)
constexpr int ROWS_T = cB * cN * cT;    // 12544
constexpr int ROWS_S = cB * cT * LS;    // 12608
constexpr int ROWS_O = cB * cS;         // 12552

// ---------------- gfx1250 async global->LDS copy (guarded) ----------------
#if defined(__has_builtin)
#if __has_builtin(__builtin_amdgcn_global_load_async_to_lds_b128) && \
    __has_builtin(__builtin_amdgcn_s_wait_asynccnt)
#define USE_ASYNC_LDS 1
#endif
#endif
#ifndef USE_ASYNC_LDS
#define USE_ASYNC_LDS 0
#endif

#if USE_ASYNC_LDS
typedef int gv4i __attribute__((vector_size(16)));
typedef __attribute__((address_space(1))) gv4i* gv4i_g;
typedef __attribute__((address_space(3))) gv4i* gv4i_l;
#endif

__device__ __forceinline__ void cp16_to_lds(__bf16* l, const __bf16* g)
{
#if USE_ASYNC_LDS
    __builtin_amdgcn_global_load_async_to_lds_b128((gv4i_g)g, (gv4i_l)l, 0, 0);
#else
    *(bf16x8*)l = *(const bf16x8*)g;
#endif
}

template <int N>
__device__ __forceinline__ void wait_async()
{
#if USE_ASYNC_LDS
    __builtin_amdgcn_s_wait_asynccnt(N);
#endif
}

// ---------------- WMMA GEMM: C = A(MxK,bf16) * Bt(NxK,bf16)^T ----------------
// double-buffered LDS tiles; async tile DMA overlapped with WMMA compute
constexpr int GBM = 128, GBN = 128, GBK = 32, GLDA = 40;

__global__ void __launch_bounds__(256)
gemm_bf16(const __bf16* __restrict__ A, const __bf16* __restrict__ Bt,
          int M, int Nn, int K,
          float* __restrict__ Cf, __bf16* __restrict__ Cb,
          const float* __restrict__ bias, const float* __restrict__ addf)
{
    __shared__ __bf16 As[2][GBM * GLDA];
    __shared__ __bf16 Bs[2][GBN * GLDA];

    const int tid  = threadIdx.x;
    const int lane = tid & 31;
    const int wave = tid >> 5;
    const int wrow = wave >> 1;       // 0..3  (32-row group)
    const int wcol = wave & 1;        // 0..1  (64-col group)
    const int rowBase = blockIdx.y * GBM;
    const int colBase = blockIdx.x * GBN;
    const int lm = lane & 15;
    const int kb = (lane >> 4) << 3;  // 0 or 8
    const int rsel = (lane >> 4) << 3;

    v8f acc[2][4];
#pragma unroll
    for (int mi = 0; mi < 2; ++mi)
#pragma unroll
        for (int ni = 0; ni < 4; ++ni)
#pragma unroll
            for (int j = 0; j < 8; ++j) acc[mi][ni][j] = 0.0f;

    // stage one 128x32 A tile + 128x32 B tile: 4 x b128 copies per thread
    auto stage = [&](int buf, int k0) {
#pragma unroll
        for (int u = 0; u < 2; ++u) {
            int c  = tid + u * 256;
            int r  = c >> 2;
            int cc = (c & 3) << 3;
            int gr = rowBase + r;
            if (gr >= M) gr = M - 1;              // clamp: keeps async count exact
            cp16_to_lds(&As[buf][r * GLDA + cc], A + (size_t)gr * K + k0 + cc);
        }
#pragma unroll
        for (int u = 0; u < 2; ++u) {
            int c  = tid + u * 256;
            int r  = c >> 2;
            int cc = (c & 3) << 3;
            int gc = colBase + r;
            if (gc >= Nn) gc = Nn - 1;
            cp16_to_lds(&Bs[buf][r * GLDA + cc], Bt + (size_t)gc * K + k0 + cc);
        }
    };

    stage(0, 0);
    const int nk = K / GBK;
    for (int ki = 0; ki < nk; ++ki) {
        const int cur = ki & 1;
        if (ki + 1 < nk) {
            stage(cur ^ 1, (ki + 1) * GBK);   // prefetch next tile (stays in flight)
            wait_async<4>();                  // previous 4 copies (this tile) done
        } else {
            wait_async<0>();
        }
        __syncthreads();

        v16bf af[2], bfm[4];
#pragma unroll
        for (int mi = 0; mi < 2; ++mi) {
            int r = wrow * 32 + mi * 16 + lm;
            bf16x8 lo = *(const bf16x8*)&As[cur][r * GLDA + kb];
            bf16x8 hi = *(const bf16x8*)&As[cur][r * GLDA + kb + 16];
            af[mi] = __builtin_shufflevector(lo, hi, 0,1,2,3,4,5,6,7,8,9,10,11,12,13,14,15);
        }
#pragma unroll
        for (int ni = 0; ni < 4; ++ni) {
            int r = wcol * 64 + ni * 16 + lm;
            bf16x8 lo = *(const bf16x8*)&Bs[cur][r * GLDA + kb];
            bf16x8 hi = *(const bf16x8*)&Bs[cur][r * GLDA + kb + 16];
            bfm[ni] = __builtin_shufflevector(lo, hi, 0,1,2,3,4,5,6,7,8,9,10,11,12,13,14,15);
        }
#pragma unroll
        for (int mi = 0; mi < 2; ++mi)
#pragma unroll
            for (int ni = 0; ni < 4; ++ni)
                acc[mi][ni] = __builtin_amdgcn_wmma_f32_16x16x32_bf16(
                    false, af[mi], false, bfm[ni], (short)0, acc[mi][ni], false, false);
        __syncthreads();   // all waves done reading 'cur' before it is overwritten
    }

#pragma unroll
    for (int mi = 0; mi < 2; ++mi) {
#pragma unroll
        for (int ni = 0; ni < 4; ++ni) {
            int col = colBase + wcol * 64 + ni * 16 + lm;
#pragma unroll
            for (int j = 0; j < 8; ++j) {
                int row = rowBase + wrow * 32 + mi * 16 + rsel + j;
                if (row < M && col < Nn) {
                    float v = acc[mi][ni][j];
                    if (bias) v += bias[col];
                    if (addf) v += addf[(size_t)row * Nn + col];
                    if (Cf) Cf[(size_t)row * Nn + col] = v;
                    if (Cb) Cb[(size_t)row * Nn + col] = (__bf16)v;
                }
            }
        }
    }
}

// ---------------- weight transpose + cast: in(KxN f32) -> out(NxK bf16) ----------------
__global__ void transpose_cast(const float* __restrict__ in, __bf16* __restrict__ out,
                               int K, int Nn)
{
    size_t idx = (size_t)blockIdx.x * blockDim.x + threadIdx.x;
    if (idx >= (size_t)K * Nn) return;
    int n = (int)(idx % Nn);
    int k = (int)(idx / Nn);
    out[(size_t)n * K + k] = (__bf16)in[idx];
}

// ---------------- build xt (B*N, T, D) bf16 from x ----------------
__global__ void build_xt(const float* __restrict__ x, __bf16* __restrict__ xt)
{
    size_t idx = (size_t)blockIdx.x * blockDim.x + threadIdx.x;
    if (idx >= (size_t)ROWS_T * cD) return;
    int d = (int)(idx % cD);
    size_t r = idx / cD;
    int t = (int)(r % cT);
    size_t m = r / cT;
    int n = (int)(m % cN);
    int b = (int)(m / cN);
    xt[idx] = (__bf16)x[((size_t)b * cS + 1 + (size_t)t * cN + n) * cD + d];
}

// ---------------- temporal attention: seq len 8, one block per m ----------------
__global__ void __launch_bounds__(128)
attn_temporal(const __bf16* __restrict__ qkv, __bf16* __restrict__ outp)
{
    __shared__ __bf16 sh[cT * D3];    // 36.9 KB
    const int m = blockIdx.x;
    const __bf16* src = qkv + (size_t)m * cT * D3;
    for (int c = threadIdx.x; c < (cT * D3) / 8; c += 128)
        *(bf16x8*)&sh[c * 8] = *(const bf16x8*)&src[c * 8];
    __syncthreads();

    const int h = threadIdx.x >> 3;
    const int i = threadIdx.x & 7;
    if (h >= cH) return;

    const __bf16* qr = &sh[i * D3 + h * cDH];
    float s[cT];
    float mx = -3.0e38f;
#pragma unroll
    for (int j = 0; j < cT; ++j) {
        const __bf16* kr = &sh[j * D3 + cD + h * cDH];
        float d = 0.0f;
#pragma unroll
        for (int e = 0; e < cDH; ++e) d += (float)qr[e] * (float)kr[e];
        s[j] = d * 0.125f;
        mx = fmaxf(mx, s[j]);
    }
    float sum = 0.0f;
#pragma unroll
    for (int j = 0; j < cT; ++j) { s[j] = __expf(s[j] - mx); sum += s[j]; }
    const float inv = 1.0f / sum;

    __bf16* orow = outp + ((size_t)m * cT + i) * cD + h * cDH;
#pragma unroll 8
    for (int e = 0; e < cDH; ++e) {
        float o = 0.0f;
#pragma unroll
        for (int j = 0; j < cT; ++j) o += s[j] * (float)sh[j * D3 + 2 * cD + h * cDH + e];
        orow[e] = (__bf16)(o * inv);
    }
}

// ---------------- build xs: cls broadcast + yt rearrange -> bf16 ----------------
__global__ void build_xs(const float* __restrict__ yt, const float* __restrict__ x,
                         __bf16* __restrict__ xs)
{
    size_t idx = (size_t)blockIdx.x * blockDim.x + threadIdx.x;
    if (idx >= (size_t)ROWS_S * cD) return;
    int d = (int)(idx % cD);
    size_t r = idx / cD;
    int j  = (int)(r % LS);
    int bt = (int)(r / LS);
    int b = bt / cT, t = bt % cT;
    float v;
    if (j == 0) {
        v = x[((size_t)b * cS) * cD + d];
    } else {
        int n = j - 1;
        size_t m = (size_t)b * cN + n;
        v = yt[(m * cT + t) * cD + d];
    }
    xs[idx] = (__bf16)v;
}

// ---------------- spatial attention (WMMA): one block per (bt, h) ----------------
// S = Q*K^T (197x197, padded 208 cols), softmax in C-registers, O = P*V (K padded 224)
constexpr int KLD  = 72;    // K rows LDS stride (bf16)
constexpr int VLD  = 232;   // Vt rows LDS stride (bf16)  (116 dwords, conflict-free frags)
constexpr int PLD  = 232;   // P rows LDS stride
constexpr int SQT  = 13;    // score col-tiles (13*16 = 208 >= 197)
constexpr int KPAD = 224;   // key dim padded to 7 K-steps of 32

__global__ void __launch_bounds__(256)
attn_spatial_wmma(const __bf16* __restrict__ qkv, __bf16* __restrict__ outp)
{
    __shared__ __bf16 Kls[208 * KLD];        // 30.0 KB : key rows (Bt layout for S)
    __shared__ __bf16 Vt[cDH * VLD];         // 29.7 KB : V transposed (Bt layout for O)
    __shared__ __bf16 Pbuf[128 * PLD];       // 59.4 KB : per-wave 16-row P tiles

    const int h  = blockIdx.x;
    const int bt = blockIdx.y;
    const __bf16* src = qkv + (size_t)bt * LS * D3;
    const int tid = threadIdx.x, lane = tid & 31, wave = tid >> 5;
    const int lm = lane & 15;
    const int kb = (lane >> 4) << 3;
    const int rsel = (lane >> 4) << 3;

    // ---- zero pad regions (keys >= 197 must contribute exact zeros) ----
    for (int i = tid; i < (208 - LS) * KLD; i += 256) Kls[LS * KLD + i] = (__bf16)0.0f;
    for (int i = tid; i < cDH * (KPAD - LS); i += 256) {
        int d = i / (KPAD - LS), k = LS + i % (KPAD - LS);
        Vt[d * VLD + k] = (__bf16)0.0f;
    }
    for (int i = tid; i < 128 * (KPAD - SQT * 16); i += 256) {
        int r = i >> 4, c = SQT * 16 + (i & 15);
        Pbuf[r * PLD + c] = (__bf16)0.0f;
    }
    // ---- stage K (rows as-is) and V (transposed) ----
    for (int c = tid; c < LS * (cDH / 8); c += 256) {
        int k = c >> 3, d8 = (c & 7) << 3;
        *(bf16x8*)&Kls[k * KLD + d8] = *(const bf16x8*)&src[(size_t)k * D3 + cD + h * cDH + d8];
    }
    for (int i = tid; i < LS * cDH; i += 256) {
        int k = i >> 6, d = i & 63;
        Vt[d * VLD + k] = src[(size_t)k * D3 + 2 * cD + h * cDH + d];
    }
    __syncthreads();

    __bf16* Pw = &Pbuf[wave * 16 * PLD];

    for (int rt = wave; rt < SQT; rt += 8) {       // wave-uniform row-tile loop
        // ---- Q fragments straight from global (2 K-steps of 32) ----
        int qrow = rt * 16 + lm;
        int qcl  = qrow < LS ? qrow : LS - 1;      // clamp; padded rows discarded later
        const __bf16* qbase = src + (size_t)qcl * D3 + h * cDH;
        v16bf aq[2];
#pragma unroll
        for (int ks = 0; ks < 2; ++ks) {
            bf16x8 lo = *(const bf16x8*)(qbase + ks * 32 + kb);
            bf16x8 hi = *(const bf16x8*)(qbase + ks * 32 + kb + 16);
            aq[ks] = __builtin_shufflevector(lo, hi, 0,1,2,3,4,5,6,7,8,9,10,11,12,13,14,15);
        }
        // ---- S tile row-block: 16 x 208 in 13 accumulators ----
        v8f sacc[SQT];
#pragma unroll
        for (int nt = 0; nt < SQT; ++nt)
#pragma unroll
            for (int j = 0; j < 8; ++j) sacc[nt][j] = 0.0f;
#pragma unroll
        for (int nt = 0; nt < SQT; ++nt) {
            int kr = nt * 16 + lm;
#pragma unroll
            for (int ks = 0; ks < 2; ++ks) {
                bf16x8 lo = *(const bf16x8*)&Kls[kr * KLD + ks * 32 + kb];
                bf16x8 hi = *(const bf16x8*)&Kls[kr * KLD + ks * 32 + kb + 16];
                v16bf bv = __builtin_shufflevector(lo, hi, 0,1,2,3,4,5,6,7,8,9,10,11,12,13,14,15);
                sacc[nt] = __builtin_amdgcn_wmma_f32_16x16x32_bf16(
                    false, aq[ks], false, bv, (short)0, sacc[nt], false, false);
            }
        }
        // ---- softmax per row (row = rsel + j; cols spread over nt x lanes 0-15) ----
#pragma unroll
        for (int j = 0; j < 8; ++j) {
            float mx = -3.0e38f;
#pragma unroll
            for (int nt = 0; nt < SQT; ++nt) {
                int col = nt * 16 + lm;
                float v = (col < LS) ? sacc[nt][j] * 0.125f : -3.0e38f;
                sacc[nt][j] = v;
                mx = fmaxf(mx, v);
            }
#pragma unroll
            for (int off = 1; off <= 8; off <<= 1) mx = fmaxf(mx, __shfl_xor(mx, off, 32));
            float sum = 0.0f;
#pragma unroll
            for (int nt = 0; nt < SQT; ++nt) {
                float p = __expf(sacc[nt][j] - mx);
                sacc[nt][j] = p;
                sum += p;
            }
#pragma unroll
            for (int off = 1; off <= 8; off <<= 1) sum += __shfl_xor(sum, off, 32);
            float inv = 1.0f / sum;
#pragma unroll
            for (int nt = 0; nt < SQT; ++nt) sacc[nt][j] *= inv;
        }
        // ---- P tile -> per-wave LDS (bf16, A-operand layout source) ----
#pragma unroll
        for (int j = 0; j < 8; ++j) {
            int rl = rsel + j;
#pragma unroll
            for (int nt = 0; nt < SQT; ++nt)
                Pw[rl * PLD + nt * 16 + lm] = (__bf16)sacc[nt][j];
        }
        __builtin_amdgcn_wave_barrier();
        // ---- O tile = P(16x224) @ V(224x64) ----
        v8f oacc[4];
#pragma unroll
        for (int ni = 0; ni < 4; ++ni)
#pragma unroll
            for (int j = 0; j < 8; ++j) oacc[ni][j] = 0.0f;
#pragma unroll
        for (int ks = 0; ks < KPAD / 32; ++ks) {
            bf16x8 lo = *(const bf16x8*)&Pw[lm * PLD + ks * 32 + kb];
            bf16x8 hi = *(const bf16x8*)&Pw[lm * PLD + ks * 32 + kb + 16];
            v16bf ap = __builtin_shufflevector(lo, hi, 0,1,2,3,4,5,6,7,8,9,10,11,12,13,14,15);
#pragma unroll
            for (int ni = 0; ni < 4; ++ni) {
                int vr = ni * 16 + lm;
                bf16x8 blo = *(const bf16x8*)&Vt[vr * VLD + ks * 32 + kb];
                bf16x8 bhi = *(const bf16x8*)&Vt[vr * VLD + ks * 32 + kb + 16];
                v16bf bv = __builtin_shufflevector(blo, bhi, 0,1,2,3,4,5,6,7,8,9,10,11,12,13,14,15);
                oacc[ni] = __builtin_amdgcn_wmma_f32_16x16x32_bf16(
                    false, ap, false, bv, (short)0, oacc[ni], false, false);
            }
        }
        // ---- store O rows < 197 ----
#pragma unroll
        for (int ni = 0; ni < 4; ++ni) {
#pragma unroll
            for (int j = 0; j < 8; ++j) {
                int row = rt * 16 + rsel + j;
                if (row < LS)
                    outp[((size_t)bt * LS + row) * cD + h * cDH + ni * 16 + lm] =
                        (__bf16)oacc[ni][j];
            }
        }
        __builtin_amdgcn_wave_barrier();   // keep Pw write/read ordering across iterations
    }
}

// ---------------- assemble attn output + residual: res = attn + x ----------------
__global__ void assemble_res(const float* __restrict__ ys, const float* __restrict__ x,
                             float* __restrict__ res)
{
    size_t idx = (size_t)blockIdx.x * blockDim.x + threadIdx.x;
    if (idx >= (size_t)ROWS_O * cD) return;
    int d = (int)(idx % cD);
    size_t rr = idx / cD;
    int s = (int)(rr % cS);
    int b = (int)(rr / cS);
    float a;
    if (s == 0) {
        a = 0.0f;
#pragma unroll
        for (int t = 0; t < cT; ++t)
            a += ys[((size_t)(b * cT + t) * LS) * cD + d];
        a *= 0.125f;
    } else {
        int q = s - 1;
        int t = q / cN, n = q % cN;
        a = ys[((size_t)(b * cT + t) * LS + 1 + n) * cD + d];
    }
    res[idx] = a + x[idx];
}

// ---------------- LayerNorm over (S,D) per batch : two-pass, deterministic ----------------
constexpr int LN_BLKS = 32;
constexpr int LN_CNT  = cS * cD;            // 1,204,992
constexpr int LN_CHUNK = LN_CNT / LN_BLKS;  // 37,656

__global__ void __launch_bounds__(256)
ln_partial(const float* __restrict__ res, float* __restrict__ part)
{
    __shared__ float ssum[256];
    __shared__ float ssq[256];
    const int b = blockIdx.y, blk = blockIdx.x;
    const float* p = res + (size_t)b * LN_CNT + (size_t)blk * LN_CHUNK;
    float s = 0.0f, q = 0.0f;
    for (int i = threadIdx.x; i < LN_CHUNK; i += 256) {
        float v = p[i];
        s += v; q += v * v;
    }
    ssum[threadIdx.x] = s; ssq[threadIdx.x] = q;
    __syncthreads();
    for (int off = 128; off > 0; off >>= 1) {
        if (threadIdx.x < off) {
            ssum[threadIdx.x] += ssum[threadIdx.x + off];
            ssq[threadIdx.x]  += ssq[threadIdx.x + off];
        }
        __syncthreads();
    }
    if (threadIdx.x == 0) {
        part[(b * LN_BLKS + blk) * 2 + 0] = ssum[0];
        part[(b * LN_BLKS + blk) * 2 + 1] = ssq[0];
    }
}

__global__ void ln_finalize(const float* __restrict__ part, float* __restrict__ stats)
{
    int b = threadIdx.x;
    if (b >= cB) return;
    float s = 0.0f, q = 0.0f;
    for (int i = 0; i < LN_BLKS; ++i) {
        s += part[(b * LN_BLKS + i) * 2 + 0];
        q += part[(b * LN_BLKS + i) * 2 + 1];
    }
    const float invc = 1.0f / (float)LN_CNT;
    float mean = s * invc;
    float var  = q * invc - mean * mean;
    stats[2 * b + 0] = mean;
    stats[2 * b + 1] = rsqrtf(var + 1e-5f);
}

__global__ void apply_ln(const float* __restrict__ res, const float* __restrict__ stats,
                         const float* __restrict__ gamma, const float* __restrict__ beta,
                         __bf16* __restrict__ normed)
{
    size_t idx = (size_t)blockIdx.x * blockDim.x + threadIdx.x;
    if (idx >= (size_t)ROWS_O * cD) return;
    size_t sd = idx % ((size_t)cS * cD);
    int b = (int)(idx / ((size_t)cS * cD));
    float mean = stats[2 * b + 0], rstd = stats[2 * b + 1];
    normed[idx] = (__bf16)((res[idx] - mean) * rstd * gamma[sd] + beta[sd]);
}

// ---------------- launcher ----------------
static inline unsigned ceil_div_u(size_t a, unsigned b) { return (unsigned)((a + b - 1) / b); }

extern "C" void kernel_launch(void* const* d_in, const int* in_sizes, int n_in,
                              void* d_out, int out_size, void* d_ws, size_t ws_size,
                              hipStream_t stream)
{
    const float* x      = (const float*)d_in[0];
    const float* qkv_t  = (const float*)d_in[1];
    const float* proj_t = (const float*)d_in[2];
    const float* qkv_s  = (const float*)d_in[3];
    const float* proj_s = (const float*)d_in[4];
    const float* gamma  = (const float*)d_in[5];
    const float* beta   = (const float*)d_in[6];
    const float* mlp_w  = (const float*)d_in[7];
    const float* mlp_b  = (const float*)d_in[8];

    char* ws = (char*)d_ws;
    size_t off = 0;
    auto take = [&](size_t bytes) { char* p = ws + off; off += (bytes + 255) & ~size_t(255); return p; };

    __bf16* w_qkv_t = (__bf16*)take((size_t)D3 * cD * 2);
    __bf16* w_prj_t = (__bf16*)take((size_t)cD * cD * 2);
    __bf16* w_qkv_s = (__bf16*)take((size_t)D3 * cD * 2);
    __bf16* w_prj_s = (__bf16*)take((size_t)cD * cD * 2);
    __bf16* w_mlp   = (__bf16*)take((size_t)cD * cD * 2);
    char* buf1 = take((size_t)ROWS_S * cD * 2);
    char* buf2 = take((size_t)ROWS_S * D3 * 2);
    char* buf3 = take((size_t)ROWS_S * cD * 2);
    char* buf4 = take((size_t)ROWS_O * cD * 4);
    float* part  = (float*)take((size_t)cB * LN_BLKS * 2 * 4);
    float* stats = (float*)take((size_t)cB * 2 * 4);

    // 1) weights -> bf16 transposed
    transpose_cast<<<ceil_div_u((size_t)cD * D3, 256), 256, 0, stream>>>(qkv_t,  w_qkv_t, cD, D3);
    transpose_cast<<<ceil_div_u((size_t)cD * cD, 256), 256, 0, stream>>>(proj_t, w_prj_t, cD, cD);
    transpose_cast<<<ceil_div_u((size_t)cD * D3, 256), 256, 0, stream>>>(qkv_s,  w_qkv_s, cD, D3);
    transpose_cast<<<ceil_div_u((size_t)cD * cD, 256), 256, 0, stream>>>(proj_s, w_prj_s, cD, cD);
    transpose_cast<<<ceil_div_u((size_t)cD * cD, 256), 256, 0, stream>>>(mlp_w,  w_mlp,  cD, cD);

    // 2) temporal path
    __bf16* xt = (__bf16*)buf1;
    build_xt<<<ceil_div_u((size_t)ROWS_T * cD, 256), 256, 0, stream>>>(x, xt);

    __bf16* qkvt = (__bf16*)buf2;
    gemm_bf16<<<dim3(D3 / GBN, (ROWS_T + GBM - 1) / GBM), 256, 0, stream>>>(
        xt, w_qkv_t, ROWS_T, D3, cD, nullptr, qkvt, nullptr, nullptr);

    __bf16* at_o = (__bf16*)buf1;
    attn_temporal<<<ROWS_T / cT, 128, 0, stream>>>(qkvt, at_o);

    float* yt = (float*)buf2;
    gemm_bf16<<<dim3(cD / GBN, (ROWS_T + GBM - 1) / GBM), 256, 0, stream>>>(
        at_o, w_prj_t, ROWS_T, cD, cD, yt, nullptr, nullptr, nullptr);

    // 3) spatial path
    __bf16* xs = (__bf16*)buf3;
    build_xs<<<ceil_div_u((size_t)ROWS_S * cD, 256), 256, 0, stream>>>(yt, x, xs);

    __bf16* qkvs = (__bf16*)buf2;
    gemm_bf16<<<dim3(D3 / GBN, (ROWS_S + GBM - 1) / GBM), 256, 0, stream>>>(
        xs, w_qkv_s, ROWS_S, D3, cD, nullptr, qkvs, nullptr, nullptr);

    __bf16* as_o = (__bf16*)buf1;
    attn_spatial_wmma<<<dim3(cH, cB * cT), 256, 0, stream>>>(qkvs, as_o);

    float* ysf = (float*)buf2;
    gemm_bf16<<<dim3(cD / GBN, (ROWS_S + GBM - 1) / GBM), 256, 0, stream>>>(
        as_o, w_prj_s, ROWS_S, cD, cD, ysf, nullptr, nullptr, nullptr);

    // 4) residual + LayerNorm
    float* res = (float*)buf4;
    assemble_res<<<ceil_div_u((size_t)ROWS_O * cD, 256), 256, 0, stream>>>(ysf, x, res);

    ln_partial<<<dim3(LN_BLKS, cB), 256, 0, stream>>>(res, part);
    ln_finalize<<<1, 32, 0, stream>>>(part, stats);

    __bf16* normed = (__bf16*)buf3;
    apply_ln<<<ceil_div_u((size_t)ROWS_O * cD, 256), 256, 0, stream>>>(res, stats, gamma, beta, normed);

    // 5) fused MLP + bias + residual -> d_out (f32)
    gemm_bf16<<<dim3(cD / GBN, (ROWS_O + GBM - 1) / GBM), 256, 0, stream>>>(
        normed, w_mlp, ROWS_O, cD, cD, (float*)d_out, nullptr, mlp_b, res);

    (void)in_sizes; (void)n_in; (void)out_size; (void)ws_size;
}
